// RWKV_21457656610848
// MI455X (gfx1250) — compile-verified, matrix operand order checked by reading, not run on hardware
//
#include <hip/hip_runtime.h>
#include <cstdint>

#define B_ 2
#define T_ 1024
#define C_ 2048
#define H_ 32
#define NH 64
#define BT (B_*T_)
#define DD 96
#define DA 96
#define DV 64
#define DG 256

typedef __attribute__((ext_vector_type(16))) __bf16 v16bf;
typedef __attribute__((ext_vector_type(8)))  float  v8f;

union FragU { v16bf v; uint4 q[2]; };

__device__ __forceinline__ unsigned short f2bf(float f) {
  union { float f; unsigned u; } x; x.f = f;
  unsigned u = x.u;
  return (unsigned short)((u + 0x7FFFu + ((u >> 16) & 1u)) >> 16);
}

__device__ __forceinline__ float sigf(float z) { return 1.0f / (1.0f + __expf(-z)); }

// ---------------------------------------------------------------- converts
__global__ void k_convert_bf16(const float* __restrict__ in,
                               unsigned short* __restrict__ out,
                               int R, int Cc, int transpose) {
  int i = blockIdx.x * blockDim.x + threadIdx.x;
  int n = R * Cc;
  if (i >= n) return;
  int r = i / Cc, c = i - r * Cc;
  unsigned short v = f2bf(in[i]);
  if (transpose) out[(size_t)c * R + r] = v;
  else           out[i] = v;
}

// ---------------------------------------------------------------- token shift + mix
__global__ void k_mix(const float* __restrict__ x, const float* __restrict__ shift,
                      const float* __restrict__ mr, const float* __restrict__ mw,
                      const float* __restrict__ mk, const float* __restrict__ mv,
                      const float* __restrict__ ma, const float* __restrict__ mg,
                      unsigned short* __restrict__ oxr, unsigned short* __restrict__ oxw,
                      unsigned short* __restrict__ oxk, unsigned short* __restrict__ oxv,
                      unsigned short* __restrict__ oxa, unsigned short* __restrict__ oxg) {
  int i = blockIdx.x * blockDim.x + threadIdx.x;
  if (i >= BT * C_) return;
  int c = i % C_;
  int bt = i / C_;
  int t = bt % T_, b = bt / T_;
  float xc = x[i];
  float prev = (t == 0) ? shift[b * C_ + c] : x[i - C_];
  float xx = prev - xc;
  oxr[i] = f2bf(xc + xx * mr[c]);
  oxw[i] = f2bf(xc + xx * mw[c]);
  oxk[i] = f2bf(xc + xx * mk[c]);
  oxv[i] = f2bf(xc + xx * mv[c]);
  oxa[i] = f2bf(xc + xx * ma[c]);
  oxg[i] = f2bf(xc + xx * mg[c]);
}

// ---------------------------------------------------------------- WMMA fragment load
// 16 rows x 32 K bf16 tile per CDNA5 ISA: lane l<16 -> row base+l, K halves 0..7 at kb,
// 8..15 at kb+16; lane>=16 same row with +8 K offset.
__device__ __forceinline__ v16bf load_frag(const unsigned short* __restrict__ p,
                                           int ld, int row, int kb, int lane) {
  int r  = row + (lane & 15);
  int kg = (lane >> 4) << 3;
  const unsigned short* base = p + (size_t)r * ld + kb + kg;
  FragU u;
  u.q[0] = *reinterpret_cast<const uint4*>(base);
  u.q[1] = *reinterpret_cast<const uint4*>(base + 16);
  return u.v;
}

// async 16B global -> LDS copy (ASYNCcnt-tracked, CDNA5 ISA)
__device__ __forceinline__ void async_cp16(unsigned lds_byte_off,
                                           const unsigned short* g) {
  asm volatile("global_load_async_to_lds_b128 %0, %1, off"
               :: "v"(lds_byte_off), "v"(g) : "memory");
}

// ---------------------------------------------------------------- WMMA GEMM  C = A[M,K] * Bt[N,K]^T
// block tile 256x64 (8 waves, each 64x32), K-step 32, double-buffered async LDS staging.
// Per wave per K-step: 8 WMMAs vs 12 ds_load_b128 (1.5 LDS loads per WMMA).
__global__ __launch_bounds__(256)
void k_gemm_bf16(const unsigned short* __restrict__ A,
                 const unsigned short* __restrict__ Bt,
                 float* __restrict__ Cout, int M, int N, int K) {
  __shared__ __align__(16) unsigned short smA[2][256 * 32];
  __shared__ __align__(16) unsigned short smB[2][64 * 32];
  int tid  = threadIdx.x;
  int lane = tid & 31;
  int wave = tid >> 5;
  int wm = wave >> 1, wn = wave & 1;   // 4 x 2 wave grid
  int mb0 = blockIdx.y * 256;
  int nb0 = blockIdx.x * 64;
  int mt = wm * 64;                    // local row within block tile
  int nt = wn * 32;                    // local col within block tile
  bool active = (nb0 + nt) < N;        // N is always a multiple of 32 here

  unsigned baseA[2] = { (unsigned)(uintptr_t)(&smA[0][0]),
                        (unsigned)(uintptr_t)(&smA[1][0]) };
  unsigned baseB[2] = { (unsigned)(uintptr_t)(&smB[0][0]),
                        (unsigned)(uintptr_t)(&smB[1][0]) };

  int nk = K >> 5;
  auto stage = [&](int ki, int bufsel) {
    int kb = ki << 5;
    // A tile: 256x32 bf16 = 1024 x 16B chunks, 4 per thread
    #pragma unroll
    for (int c = 0; c < 4; ++c) {
      int ch   = tid + c * 256;
      int rowa = ch >> 2, cola = (ch & 3) << 3;
      async_cp16(baseA[bufsel] + (unsigned)(rowa * 32 + cola) * 2,
                 A + (size_t)(mb0 + rowa) * K + kb + cola);
    }
    // B tile: 64x32 bf16 = 256 x 16B chunks, 1 per thread (guard N edge)
    int rowb = tid >> 2, colb = (tid & 3) << 3;
    if (nb0 + rowb < N)
      async_cp16(baseB[bufsel] + (unsigned)(rowb * 32 + colb) * 2,
                 Bt + (size_t)(nb0 + rowb) * K + kb + colb);
  };

  v8f acc[4][2];
  #pragma unroll
  for (int i = 0; i < 4; ++i)
    #pragma unroll
    for (int j = 0; j < 2; ++j) acc[i][j] = {};

  stage(0, 0);
  for (int i = 0; i < nk; ++i) {
    int cur = i & 1;
    if (i + 1 < nk) {
      stage(i + 1, cur ^ 1);
      asm volatile("s_wait_asynccnt 5" ::: "memory");   // stage(i) complete
    } else {
      asm volatile("s_wait_asynccnt 0" ::: "memory");
    }
    __syncthreads();
    if (active) {
      const unsigned short* pA = &smA[cur][0];
      const unsigned short* pB = &smB[cur][0];
      v16bf b0 = load_frag(pB, 32, nt,      0, lane);
      v16bf b1 = load_frag(pB, 32, nt + 16, 0, lane);
      #pragma unroll
      for (int ia = 0; ia < 4; ++ia) {
        v16bf af = load_frag(pA, 32, mt + 16 * ia, 0, lane);
        acc[ia][0] = __builtin_amdgcn_wmma_f32_16x16x32_bf16(false, af, false, b0, (short)0, acc[ia][0], false, false);
        acc[ia][1] = __builtin_amdgcn_wmma_f32_16x16x32_bf16(false, af, false, b1, (short)0, acc[ia][1], false, false);
      }
    }
    __syncthreads();   // protect buf[cur] before stage(i+2) overwrites it
  }
  if (active) {
    int coln = lane & 15;
    int rowb = (lane >> 4) * 8;
    int mg = mb0 + mt, ng = nb0 + nt;
    #pragma unroll
    for (int ia = 0; ia < 4; ++ia) {
      #pragma unroll
      for (int j = 0; j < 8; ++j) {
        Cout[(size_t)(mg + 16 * ia + rowb + j) * N + ng + coln]      = acc[ia][0][j];
        Cout[(size_t)(mg + 16 * ia + rowb + j) * N + ng + 16 + coln] = acc[ia][1][j];
      }
    }
  }
}

// ---------------------------------------------------------------- activation f32 -> bf16
__global__ void k_act(const float* __restrict__ in, unsigned short* __restrict__ out,
                      int n, int mode) {
  int i = blockIdx.x * blockDim.x + threadIdx.x;
  if (i >= n) return;
  float v = in[i];
  if (mode == 1)      v = tanhf(v);
  else if (mode == 2) v = sigf(v);
  out[i] = f2bf(v);
}

// ---------------------------------------------------------------- per-head vector prep
__global__ __launch_bounds__(64)
void k_vectors(const float* __restrict__ dw, const float* __restrict__ da,
               const float* __restrict__ dv, float* __restrict__ kf,
               float* __restrict__ vf, const float* __restrict__ v_first,
               const float* __restrict__ w0, const float* __restrict__ a0,
               const float* __restrict__ v0, const float* __restrict__ k_k,
               const float* __restrict__ k_a,
               float* __restrict__ wdec, float* __restrict__ bvec,
               float* __restrict__ avec) {
  __shared__ float skk[64];
  int tid = threadIdx.x;
  int gh  = blockIdx.x;                 // (b*T + t)*H + h
  int h   = gh % H_;
  size_t i = (size_t)gh * 64 + tid;     // flat [B,T,C]
  int c   = h * 64 + tid;
  // w = -softplus(-(w0+dw)) - 0.5  =>  exp(w) = sigmoid(w0+dw) * e^{-0.5}
  float y = w0[c] + dw[i];
  wdec[i] = __expf(-0.60653066f * sigf(y));
  float a  = sigf(a0[c] + da[i]);
  float kv = kf[i];                     // original k
  float kkv = kv * k_k[c];
  skk[tid] = kkv * kkv;
  __syncthreads();
  float ss = 0.f;
  #pragma unroll
  for (int j = 0; j < 64; ++j) ss += skk[j];
  float nrm = fmaxf(sqrtf(ss), 1e-12f);
  float kkn = kkv / nrm;
  bvec[i] = -kkn;
  avec[i] = kkn * a;
  kf[i]   = kv * (1.f + (a - 1.f) * k_a[c]);
  float vp = vf[i];
  vf[i] = vp + (v_first[i] - vp) * sigf(v0[c] + dv[i]);
}

// ---------------------------------------------------------------- WKV scan
// one block per (b,h); 256 threads; 4 lanes own one state row (16 K each);
// cross-lane dots via shfl_xor; double-buffered LDS, one barrier per step.
__global__ __launch_bounds__(256)
void k_wkv(const float* __restrict__ r, const float* __restrict__ w,
           const float* __restrict__ k, const float* __restrict__ v,
           const float* __restrict__ bb, const float* __restrict__ aa,
           const float* __restrict__ S0, float* __restrict__ o,
           float* __restrict__ Sf) {
  int tid = threadIdx.x;
  int row = tid >> 2;                   // v index 0..63
  int q   = tid & 3;                    // k-quarter
  int kb0 = q * 16;
  int bh  = blockIdx.x;
  int b = bh / H_, h = bh % H_;
  __shared__ float st[2][6][64];        // [buf][r,w,k,v,b,a][elem]
  float S[16];
  const float* s0p = S0 + ((size_t)bh * 64 + row) * 64 + kb0;
  #pragma unroll
  for (int j = 0; j < 16; ++j) S[j] = s0p[j];
  size_t base = (size_t)b * T_ * C_ + (size_t)h * 64;
  float lr = 0, lw = 0, lk = 0, lv = 0, lb = 0, la = 0;
  if (tid < 64) {
    size_t i0 = base + tid;
    lr = r[i0]; lw = w[i0]; lk = k[i0]; lv = v[i0]; lb = bb[i0]; la = aa[i0];
  }
  for (int t = 0; t < T_; ++t) {
    int cur = t & 1;
    if (tid < 64) {
      st[cur][0][tid] = lr; st[cur][1][tid] = lw; st[cur][2][tid] = lk;
      st[cur][3][tid] = lv; st[cur][4][tid] = lb; st[cur][5][tid] = la;
    }
    __syncthreads();
    if (tid < 64 && t + 1 < T_) {       // hide global latency behind compute
      size_t i1 = base + (size_t)(t + 1) * C_ + tid;
      lr = r[i1]; lw = w[i1]; lk = k[i1]; lv = v[i1]; lb = bb[i1]; la = aa[i1];
      if (t + 2 < T_) {
        size_t i2 = base + (size_t)(t + 2) * C_ + tid;
        __builtin_prefetch(&r[i2], 0, 0);  __builtin_prefetch(&w[i2], 0, 0);
        __builtin_prefetch(&k[i2], 0, 0);  __builtin_prefetch(&v[i2], 0, 0);
        __builtin_prefetch(&bb[i2], 0, 0); __builtin_prefetch(&aa[i2], 0, 0);
      }
    }
    const float* pr = st[cur][0]; const float* pw = st[cur][1];
    const float* pk = st[cur][2]; const float* pv = st[cur][3];
    const float* pb = st[cur][4]; const float* pa = st[cur][5];
    float Sa = 0.f;
    #pragma unroll
    for (int j = 0; j < 16; ++j) Sa += S[j] * pa[kb0 + j];
    Sa += __shfl_xor(Sa, 1, 32);
    Sa += __shfl_xor(Sa, 2, 32);
    float vv = pv[row];
    float acc = 0.f;
    #pragma unroll
    for (int j = 0; j < 16; ++j) {
      float s = S[j] * pw[kb0 + j] + Sa * pb[kb0 + j] + vv * pk[kb0 + j];
      S[j] = s;
      acc += s * pr[kb0 + j];
    }
    acc += __shfl_xor(acc, 1, 32);
    acc += __shfl_xor(acc, 2, 32);
    if (q == 0) o[base + (size_t)t * C_ + row] = acc;
    // no trailing barrier: double-buffered, next write goes to the other buffer
  }
  float* sfp = Sf + ((size_t)bh * 64 + row) * 64 + kb0;
  #pragma unroll
  for (int j = 0; j < 16; ++j) sfp[j] = S[j];
}

// ---------------------------------------------------------------- GroupNorm + residual + gate
__global__ __launch_bounds__(64)
void k_post(const float* __restrict__ o, const float* __restrict__ r,
            const float* __restrict__ k, const float* __restrict__ v,
            const float* __restrict__ g, const float* __restrict__ r_k,
            const float* __restrict__ gn_w, const float* __restrict__ gn_b,
            unsigned short* __restrict__ gated) {
  __shared__ float s1[64], s2[64], s3[64];
  int tid = threadIdx.x;
  int gh  = blockIdx.x;
  int h   = gh % H_;
  size_t i = (size_t)gh * 64 + tid;
  int c   = h * 64 + tid;
  float ov = o[i], rv = r[i], kv = k[i];
  s1[tid] = ov;
  s2[tid] = ov * ov;
  s3[tid] = rv * kv * r_k[c];
  __syncthreads();
  float sm = 0.f, sq = 0.f, dt = 0.f;
  #pragma unroll
  for (int j = 0; j < 64; ++j) { sm += s1[j]; sq += s2[j]; dt += s3[j]; }
  float mu  = sm * (1.f / 64.f);
  float var = sq * (1.f / 64.f) - mu * mu;
  float on  = (ov - mu) * rsqrtf(var + 6.4e-4f) * gn_w[c] + gn_b[c];
  on += dt * v[i];
  gated[i] = f2bf(on * g[i]);
}

// ---------------------------------------------------------------- launch
static inline size_t algn(size_t x) { return (x + 255) & ~(size_t)255; }

extern "C" void kernel_launch(void* const* d_in, const int* in_sizes, int n_in,
                              void* d_out, int out_size, void* d_ws, size_t ws_size,
                              hipStream_t stream) {
  const float* x        = (const float*)d_in[0];
  const float* v_first  = (const float*)d_in[1];
  const float* shift    = (const float*)d_in[2];
  const float* wkv0     = (const float*)d_in[3];
  const float* x_r      = (const float*)d_in[4];
  const float* x_w      = (const float*)d_in[5];
  const float* x_k      = (const float*)d_in[6];
  const float* x_v      = (const float*)d_in[7];
  const float* x_a      = (const float*)d_in[8];
  const float* x_g      = (const float*)d_in[9];
  const float* w0       = (const float*)d_in[10];
  const float* w1       = (const float*)d_in[11];
  const float* w2       = (const float*)d_in[12];
  const float* a0       = (const float*)d_in[13];
  const float* a1       = (const float*)d_in[14];
  const float* a2       = (const float*)d_in[15];
  const float* v0       = (const float*)d_in[16];
  const float* v1       = (const float*)d_in[17];
  const float* v2       = (const float*)d_in[18];
  const float* g1       = (const float*)d_in[19];
  const float* g2       = (const float*)d_in[20];
  const float* k_k      = (const float*)d_in[21];
  const float* k_a      = (const float*)d_in[22];
  const float* r_k      = (const float*)d_in[23];
  const float* W_r      = (const float*)d_in[24];
  const float* W_k      = (const float*)d_in[25];
  const float* W_v      = (const float*)d_in[26];
  const float* W_o      = (const float*)d_in[27];
  const float* gn_w     = (const float*)d_in[28];
  const float* gn_b     = (const float*)d_in[29];

  float* out = (float*)d_out;
  float* Sf  = out + (size_t)BT * C_;

  char* wp = (char*)d_ws;
  auto alloc = [&](size_t bytes) -> void* { void* p = (void*)wp; wp += algn(bytes); return p; };
  const size_t nBTC = (size_t)BT * C_;

  unsigned short* xr  = (unsigned short*)alloc(nBTC * 2);
  unsigned short* xw  = (unsigned short*)alloc(nBTC * 2);
  unsigned short* xk  = (unsigned short*)alloc(nBTC * 2);
  unsigned short* xv  = (unsigned short*)alloc(nBTC * 2);
  unsigned short* xa  = (unsigned short*)alloc(nBTC * 2);
  unsigned short* xg  = (unsigned short*)alloc(nBTC * 2);
  unsigned short* Wrb = (unsigned short*)alloc((size_t)C_ * C_ * 2);
  unsigned short* Wkb = (unsigned short*)alloc((size_t)C_ * C_ * 2);
  unsigned short* Wvb = (unsigned short*)alloc((size_t)C_ * C_ * 2);
  unsigned short* Wob = (unsigned short*)alloc((size_t)C_ * C_ * 2);
  unsigned short* w1t = (unsigned short*)alloc((size_t)C_ * DD * 2);
  unsigned short* w2t = (unsigned short*)alloc((size_t)C_ * DD * 2);
  unsigned short* a1t = (unsigned short*)alloc((size_t)C_ * DA * 2);
  unsigned short* a2t = (unsigned short*)alloc((size_t)C_ * DA * 2);
  unsigned short* v1t = (unsigned short*)alloc((size_t)C_ * DV * 2);
  unsigned short* v2t = (unsigned short*)alloc((size_t)C_ * DV * 2);
  unsigned short* g1t = (unsigned short*)alloc((size_t)C_ * DG * 2);
  unsigned short* g2t = (unsigned short*)alloc((size_t)C_ * DG * 2);
  unsigned short* hwb = (unsigned short*)alloc((size_t)BT * DD * 2);
  unsigned short* hab = (unsigned short*)alloc((size_t)BT * DA * 2);
  unsigned short* hvb = (unsigned short*)alloc((size_t)BT * DV * 2);
  unsigned short* hgb = (unsigned short*)alloc((size_t)BT * DG * 2);
  unsigned short* gated = (unsigned short*)alloc(nBTC * 2);

  float* rbuf = (float*)alloc(nBTC * 4);
  float* kbuf = (float*)alloc(nBTC * 4);
  float* vbuf = (float*)alloc(nBTC * 4);
  float* hw   = (float*)alloc((size_t)BT * DD * 4);
  float* ha   = (float*)alloc((size_t)BT * DA * 4);
  float* hv   = (float*)alloc((size_t)BT * DV * 4);
  float* hg   = (float*)alloc((size_t)BT * DG * 4);
  float* dw   = (float*)alloc(nBTC * 4);
  float* da   = (float*)alloc(nBTC * 4);
  float* dv   = (float*)alloc(nBTC * 4);
  float* gbuf = (float*)alloc(nBTC * 4);
  float* wdec = (float*)alloc(nBTC * 4);
  float* bvec = (float*)alloc(nBTC * 4);
  float* avec = (float*)alloc(nBTC * 4);
  float* obuf = (float*)alloc(nBTC * 4);

  auto conv = [&](const float* in, unsigned short* o_, int R, int Cc, int tr) {
    int n = R * Cc;
    k_convert_bf16<<<dim3((n + 255) / 256), dim3(256), 0, stream>>>(in, o_, R, Cc, tr);
  };
  conv(W_r, Wrb, C_, C_, 0);  conv(W_k, Wkb, C_, C_, 0);
  conv(W_v, Wvb, C_, C_, 0);  conv(W_o, Wob, C_, C_, 0);
  conv(w1, w1t, C_, DD, 1);   conv(w2, w2t, DD, C_, 1);
  conv(a1, a1t, C_, DA, 1);   conv(a2, a2t, DA, C_, 1);
  conv(v1, v1t, C_, DV, 1);   conv(v2, v2t, DV, C_, 1);
  conv(g1, g1t, C_, DG, 1);   conv(g2, g2t, DG, C_, 1);

  k_mix<<<dim3((BT * C_ + 255) / 256), dim3(256), 0, stream>>>(
      x, shift, x_r, x_w, x_k, x_v, x_a, x_g, xr, xw, xk, xv, xa, xg);

  auto gemm = [&](const unsigned short* A, const unsigned short* Bt, float* Cc,
                  int M, int N, int K) {
    dim3 g((N + 63) / 64, (M + 255) / 256);
    k_gemm_bf16<<<g, dim3(256), 0, stream>>>(A, Bt, Cc, M, N, K);
  };
  gemm(xr, Wrb, rbuf, BT, C_, C_);
  gemm(xk, Wkb, kbuf, BT, C_, C_);
  gemm(xv, Wvb, vbuf, BT, C_, C_);
  gemm(xw, w1t, hw, BT, DD, C_);
  gemm(xa, a1t, ha, BT, DA, C_);
  gemm(xv, v1t, hv, BT, DV, C_);
  gemm(xg, g1t, hg, BT, DG, C_);

  k_act<<<dim3((BT * DD + 255) / 256), dim3(256), 0, stream>>>(hw, hwb, BT * DD, 1);
  k_act<<<dim3((BT * DA + 255) / 256), dim3(256), 0, stream>>>(ha, hab, BT * DA, 0);
  k_act<<<dim3((BT * DV + 255) / 256), dim3(256), 0, stream>>>(hv, hvb, BT * DV, 0);
  k_act<<<dim3((BT * DG + 255) / 256), dim3(256), 0, stream>>>(hg, hgb, BT * DG, 2);

  gemm(hwb, w2t, dw, BT, C_, DD);
  gemm(hab, a2t, da, BT, C_, DA);
  gemm(hvb, v2t, dv, BT, C_, DV);
  gemm(hgb, g2t, gbuf, BT, C_, DG);

  k_vectors<<<dim3(BT * H_), dim3(64), 0, stream>>>(
      dw, da, dv, kbuf, vbuf, v_first, w0, a0, v0, k_k, k_a, wdec, bvec, avec);

  k_wkv<<<dim3(B_ * H_), dim3(256), 0, stream>>>(
      rbuf, wdec, kbuf, vbuf, bvec, avec, wkv0, obuf, Sf);

  k_post<<<dim3(BT * H_), dim3(64), 0, stream>>>(
      obuf, rbuf, kbuf, vbuf, gbuf, r_k, gn_w, gn_b, gated);

  gemm(gated, Wob, out, BT, C_, C_);
}
